// MNISTTriXLayer_5162550690213
// MI455X (gfx1250) — matrix operand order repaired and checked by reading, not compiled, not run on hardware
//
#include <hip/hip_runtime.h>
#include <hip/hip_bf16.h>

// Problem sizes (fixed by reference)
#define BB   32768
#define DIN  784
#define DD   256
#define TT   16
#define CC   64

// ---- WMMA / TDM types (gfx1250, wave32) ----
typedef __bf16 bf16_t;
typedef bf16_t v16bf __attribute__((ext_vector_type(16)));
typedef float  v8f   __attribute__((ext_vector_type(8)));
typedef unsigned int v4u  __attribute__((ext_vector_type(4)));
typedef int          i4   __attribute__((ext_vector_type(4)));
typedef int          i8   __attribute__((ext_vector_type(8)));

union FragU { v16bf v; v4u q[2]; };

__device__ __forceinline__ unsigned short f2bf(float f) {
    union { float f; unsigned u; } v; v.f = f;
    unsigned u = v.u;
    unsigned r = u + 0x7FFFu + ((u >> 16) & 1u);   // round-to-nearest-even
    return (unsigned short)(r >> 16);
}
__device__ __forceinline__ unsigned pack2bf(float a, float b) {
    return (unsigned)f2bf(a) | ((unsigned)f2bf(b) << 16);
}

// ---- TDM: load a [256 rows x 32 bf16] tile of Wp_t into LDS, inserting ----
// ---- 4 DWORDs of padding every 16 DWORDs (=> 80B LDS row stride).       ----
__device__ __forceinline__ void tdm_load_b(unsigned lds_addr,
                                           const unsigned short* gptr) {
    unsigned long long ga = (unsigned long long)(uintptr_t)gptr;
    v4u g0 = { 1u,                                   // count=1, no gather
               lds_addr,                             // LDS dest (bytes)
               (unsigned)(ga & 0xFFFFFFFFu),         // global_addr lo
               (unsigned)((ga >> 32) & 0x01FFFFFFu) | (2u << 30) }; // hi | type=2
    i8 g1 = { (int)0x06D10000,  // data_size=2B, pad_en, pad_interval=16dw, pad_amt=4dw
              (int)0x10000000,  // tensor_dim0 = 4096 (no OOB in-tile)
              (int)0x10000000,  // tensor_dim1 = 4096
              (int)0x00200000,  // tile_dim0 = 32 elements
              (int)0x00000100,  // tile_dim1 = 256 rows
              (int)800,         // tensor_dim0_stride = 800 elements
              0, 0 };
    i4 z4 = { 0, 0, 0, 0 };
    i8 z8 = { 0, 0, 0, 0, 0, 0, 0, 0 };
    __builtin_amdgcn_tensor_load_to_lds(g0, g1, z4, z4, z8, 0);
}

// -------------------- Kernel P1: Wp -> Wp_t[256][800] bf16 (zero-padded K) --
__global__ void kprep_wp(const float* __restrict__ Wp, unsigned short* __restrict__ Wp_t) {
    int idx = blockIdx.x * 256 + threadIdx.x;          // 256*800 total
    int n = idx / 800, k = idx - n * 800;
    float v = (k < DIN) ? Wp[(size_t)k * DD + n] : 0.0f;
    Wp_t[idx] = f2bf(v);
}

// -------------------- Kernel P2: Wc -> Wc_t[16][64][256] bf16 (N-major) ----
__global__ void kprep_wc(const float* __restrict__ Wc, unsigned short* __restrict__ Wc_t) {
    int idx = blockIdx.x * 256 + threadIdx.x;          // 16*64*256 total
    int k = idx & 255, n = (idx >> 8) & 63, t = idx >> 14;
    Wc_t[idx] = f2bf(Wc[((size_t)t * DD + k) * CC + n]);
}

// -------------------- Kernel 0: zero bucket counters ------------------------
__global__ void k0_init(int* counts) {
    if (threadIdx.x < TT) counts[threadIdx.x] = 0;
}

// -------------------- Kernel 1: h = x@Wp + bp, WMMA scores, argmax, buckets -
// grid: B/64 = 512 blocks, 256 threads (8 waves). Block: 64 rows x 256 cols.
// B-tiles double-buffered via TDM (wave 0); A-tiles double-buffered via regs.
// Routing scores also computed with WMMA (ternary sigs are exact in bf16).
__global__ void k1_gemm_route(const float* __restrict__ x,
                              const unsigned short* __restrict__ Wp_t,
                              const float* __restrict__ bp,
                              const float* __restrict__ sigs_raw,
                              int* __restrict__ counts,
                              int* __restrict__ bucket,
                              unsigned int* __restrict__ h_ws,
                              int* __restrict__ tidx_out) {
    __shared__ unsigned short Asub[2][64][40];   // 2 x (rows x 32 bf16 + 8 pad)
    __shared__ unsigned short Bsub[2][256][40];  // 2 x (N-major x 32 bf16 + 8 pad)
    __shared__ unsigned short Hs[64][256];       // h block, bf16
    __shared__ unsigned short Sb[TT][264];       // ternary sigs, bf16, N-major
    __shared__ float          Sc[2][64][TT];     // partial score tiles (K halves)

    const int tid  = threadIdx.x;
    const int lane = tid & 31;
    const int wid  = tid >> 5;
    const int mtile = wid & 3;
    const int nbase = (wid >> 2) * 8;
    const int half  = lane >> 4;
    const int lm    = lane & 15;
    const int row0  = blockIdx.x * 64;

    v8f acc[8];
#pragma unroll
    for (int i = 0; i < 8; ++i) acc[i] = (v8f)(0.0f);

    // A-staging coords: thread loads 8 consecutive k for one row
    const int sr = tid >> 2;            // 0..63
    const int sk = (tid & 3) * 8;       // 0,8,16,24
    const float* xrow = x + (size_t)(row0 + sr) * DIN + sk;

    float ar[8];
    // ---- prologue: chunk 0 ----
    {
        float4 a = *(const float4*)(xrow);
        float4 b = *(const float4*)(xrow + 4);
        ar[0]=a.x; ar[1]=a.y; ar[2]=a.z; ar[3]=a.w;
        ar[4]=b.x; ar[5]=b.y; ar[6]=b.z; ar[7]=b.w;
        unsigned* dst = (unsigned*)&Asub[0][sr][sk];
        dst[0]=pack2bf(ar[0],ar[1]); dst[1]=pack2bf(ar[2],ar[3]);
        dst[2]=pack2bf(ar[4],ar[5]); dst[3]=pack2bf(ar[6],ar[7]);
        if (wid == 0)
            tdm_load_b((unsigned)(uintptr_t)&Bsub[0][0][0], Wp_t);
    }

    for (int i = 0; i < 25; ++i) {
        if (wid == 0) __builtin_amdgcn_s_wait_tensorcnt(0);
        __syncthreads();
        const int buf = i & 1;
        const bool haveNext = (i < 24);

        // issue next chunk's global loads early (latency hidden by WMMA)
        if (haveNext) {
            const int k0 = (i + 1) * 32;
            if (k0 + sk + 8 <= DIN) {               // whole octet valid
                float4 a = *(const float4*)(xrow + k0);
                float4 b = *(const float4*)(xrow + k0 + 4);
                ar[0]=a.x; ar[1]=a.y; ar[2]=a.z; ar[3]=a.w;
                ar[4]=b.x; ar[5]=b.y; ar[6]=b.z; ar[7]=b.w;
            } else {                                 // whole octet padded
#pragma unroll
                for (int j = 0; j < 8; ++j) ar[j] = 0.0f;
            }
            if (wid == 0)
                tdm_load_b((unsigned)(uintptr_t)&Bsub[buf ^ 1][0][0],
                           Wp_t + k0);
        }

        // ---- compute from buf ----
        FragU a;
        {
            const char* ap = (const char*)&Asub[buf][16 * mtile + lm][0];
            a.q[0] = *(const v4u*)(ap + half * 16);
            a.q[1] = *(const v4u*)(ap + half * 16 + 32);
        }
        FragU bfr[8];
#pragma unroll
        for (int j = 0; j < 8; ++j) {
            const char* bpr = (const char*)&Bsub[buf][16 * (nbase + j) + lm][0];
            bfr[j].q[0] = *(const v4u*)(bpr + half * 16);
            bfr[j].q[1] = *(const v4u*)(bpr + half * 16 + 32);
        }
#pragma unroll
        for (int j = 0; j < 8; ++j)
            acc[j] = __builtin_amdgcn_wmma_f32_16x16x32_bf16(
                false, a.v, false, bfr[j].v, (short)0, acc[j], false, false);

        // ---- store next A chunk into the other buffer ----
        if (haveNext) {
            unsigned* dst = (unsigned*)&Asub[buf ^ 1][sr][sk];
            dst[0]=pack2bf(ar[0],ar[1]); dst[1]=pack2bf(ar[2],ar[3]);
            dst[2]=pack2bf(ar[4],ar[5]); dst[3]=pack2bf(ar[6],ar[7]);
        }
    }

    // ---- accumulators (+bp) -> Hs (bf16) ----
#pragma unroll
    for (int i = 0; i < 8; ++i) {
        const int n = 16 * (nbase + i) + lm;
        const float bpv = bp[n];
#pragma unroll
        for (int r = 0; r < 8; ++r)
            Hs[16 * mtile + r + 8 * half][n] = f2bf(acc[i][r] + bpv);
    }
    // ---- stage ternary signatures as bf16 (exact) ----
#pragma unroll
    for (int j = 0; j < 16; ++j) {
        int e = tid + j * 256;               // e = t*256 + d
        float s = sigs_raw[e];
        float q = (s > 0.3f) ? 1.0f : ((s < -0.3f) ? -1.0f : 0.0f);
        Sb[e >> 8][e & 255] = f2bf(q);
    }
    __syncthreads();

    // ---- write h block to workspace (coalesced b128) ----
    {
        const v4u* src = (const v4u*)&Hs[0][0];
        v4u* dst = (v4u*)h_ws + (size_t)blockIdx.x * 2048;
#pragma unroll
        for (int j = 0; j < 8; ++j) dst[tid + j * 256] = src[tid + j * 256];
    }

    // ---- scores via WMMA: h[64x256] @ sigs^T[256x16] ----
    // wave w: score tile mtile = w&3, K-half = w>>2 (4 WMMAs each)
    {
        const int khalf = wid >> 2;
        v8f sacc = (v8f)(0.0f);
#pragma unroll
        for (int ks = 0; ks < 4; ++ks) {
            const int k0 = khalf * 128 + ks * 32;
            FragU a;
            const char* ap = (const char*)&Hs[16 * mtile + lm][0] + k0 * 2;
            a.q[0] = *(const v4u*)(ap + half * 16);
            a.q[1] = *(const v4u*)(ap + half * 16 + 32);
            FragU b;
            const char* bpr = (const char*)&Sb[lm][0] + k0 * 2;
            b.q[0] = *(const v4u*)(bpr + half * 16);
            b.q[1] = *(const v4u*)(bpr + half * 16 + 32);
            sacc = __builtin_amdgcn_wmma_f32_16x16x32_bf16(
                false, a.v, false, b.v, (short)0, sacc, false, false);
        }
#pragma unroll
        for (int r = 0; r < 8; ++r)
            Sc[khalf][16 * mtile + r + 8 * half][lm] = sacc[r];
    }
    __syncthreads();

    // ---- argmax (first-max, ascending t) + bucket append ----
    if (tid < 64) {
        float bestv = -3.0e38f; int bestt = 0;
#pragma unroll
        for (int t = 0; t < TT; ++t) {
            float s = Sc[0][tid][t] + Sc[1][tid][t];
            if (s > bestv) { bestv = s; bestt = t; }
        }
        const int grow = row0 + tid;
        tidx_out[grow] = bestt;
        int pos = atomicAdd(&counts[bestt], 1);
        bucket[bestt * BB + pos] = grow;
    }
}

// -------------------- Kernel 2: routed classifier via compacted buckets ----
// grid: (256, T). block = 256 threads = 8 waves; wave w handles one 16-row chunk.
__global__ void k2_classify(const unsigned short* __restrict__ Wc_t,
                            const float* __restrict__ bc,
                            const int* __restrict__ counts,
                            const int* __restrict__ bucket,
                            const unsigned int* __restrict__ h_ws,
                            float* __restrict__ out) {
    const int t = blockIdx.y;
    const int count = counts[t];
    const int blk_row0 = blockIdx.x * 128;
    if (blk_row0 >= count) return;       // uniform early-out

    __shared__ unsigned short Wcs[64][264];
    __shared__ unsigned short Ach[8][16][264];
    __shared__ int rows_lds[8][16];

    const int tid  = threadIdx.x;
    const int lane = tid & 31;
    const int wid  = tid >> 5;
    const int half = lane >> 4;
    const int lm   = lane & 15;

    // stage Wc_t[t] (already bf16, N-major) via b128 copies
    {
        const int n = tid & 63;
        const int seg = (tid >> 6) * 64;
        const v4u* src = (const v4u*)(Wc_t + ((size_t)t * 64 + n) * 256 + seg);
        v4u* dst = (v4u*)&Wcs[n][seg];
#pragma unroll
        for (int j = 0; j < 8; ++j) dst[j] = src[j];
    }

    const int chunk_row = blk_row0 + wid * 16;
    const bool active = chunk_row < count;
    if (active) {
        for (int r = 0; r < 16; ++r) {
            int gi = chunk_row + r; if (gi > count - 1) gi = count - 1;
            const int grow = bucket[t * BB + gi];
            rows_lds[wid][r] = grow;
            const v4u* src = (const v4u*)(h_ws + (size_t)grow * 128);
            v4u* dst = (v4u*)&Ach[wid][r][0];
            dst[lane] = src[lane];
        }
    }
    __syncthreads();

    if (active) {
        v8f acc[4];
#pragma unroll
        for (int i = 0; i < 4; ++i) acc[i] = (v8f)(0.0f);

#pragma unroll
        for (int k0 = 0; k0 < DD; k0 += 32) {
            FragU a;
            const char* ap = (const char*)&Ach[wid][lm][0] + k0 * 2;
            a.q[0] = *(const v4u*)(ap + half * 16);
            a.q[1] = *(const v4u*)(ap + half * 16 + 32);
#pragma unroll
            for (int nt = 0; nt < 4; ++nt) {
                const int n = nt * 16 + lm;
                FragU b;
                const char* bpr = (const char*)&Wcs[n][0] + k0 * 2;
                b.q[0] = *(const v4u*)(bpr + half * 16);
                b.q[1] = *(const v4u*)(bpr + half * 16 + 32);
                acc[nt] = __builtin_amdgcn_wmma_f32_16x16x32_bf16(
                    false, a.v, false, b.v, (short)0, acc[nt], false, false);
            }
        }
#pragma unroll
        for (int nt = 0; nt < 4; ++nt) {
            const int n = nt * 16 + lm;
            const float bcv = bc[t * CC + n];
#pragma unroll
            for (int r = 0; r < 8; ++r) {
                const int m = r + 8 * half;
                if (chunk_row + m < count) {
                    const int row = rows_lds[wid][m];
                    out[(size_t)row * CC + n] = acc[nt][r] + bcv;
                }
            }
        }
    }
}

// ---------------------------------------------------------------------------
extern "C" void kernel_launch(void* const* d_in, const int* in_sizes, int n_in,
                              void* d_out, int out_size, void* d_ws, size_t ws_size,
                              hipStream_t stream) {
    const float* x        = (const float*)d_in[0];
    const float* Wp       = (const float*)d_in[1];
    const float* bp       = (const float*)d_in[2];
    const float* sigs_raw = (const float*)d_in[3];
    const float* Wc       = (const float*)d_in[4];
    const float* bc       = (const float*)d_in[5];

    float* logits = (float*)d_out;                            // [B, C]
    int*   tidx   = (int*)((float*)d_out + (size_t)BB * CC);  // [B]

    // ws: counters(1KB) | buckets(2MB) | h bf16(16MB) | Wp_t(400KB) | Wc_t(512KB)
    char* ws = (char*)d_ws;
    int* counts = (int*)ws;
    int* bucket = (int*)(ws + 1024);
    unsigned int* h_ws = (unsigned int*)(ws + 1024 + (size_t)TT * BB * 4);
    unsigned short* Wp_t = (unsigned short*)(ws + 1024 + (size_t)TT * BB * 4
                                             + (size_t)BB * DD * 2);
    unsigned short* Wc_t = (unsigned short*)((char*)Wp_t + (size_t)DD * 800 * 2);

    kprep_wp<<<800, 256, 0, stream>>>(Wp, Wp_t);
    kprep_wc<<<1024, 256, 0, stream>>>(Wc, Wc_t);
    k0_init<<<1, 32, 0, stream>>>(counts);
    k1_gemm_route<<<BB / 64, 256, 0, stream>>>(x, Wp_t, bp, sigs_raw,
                                               counts, bucket, h_ws, tidx);
    k2_classify<<<dim3(256, TT), 256, 0, stream>>>(Wc_t, bc, counts, bucket,
                                                   h_ws, logits);
}
